// MXFP4Linear_50139448213871
// MI455X (gfx1250) — matrix operand order repaired and checked by reading, not compiled, not run on hardware
//
#include <hip/hip_runtime.h>

typedef __attribute__((ext_vector_type(8)))  __bf16 v8bf;
typedef __attribute__((ext_vector_type(16))) __bf16 v16bf;
typedef __attribute__((ext_vector_type(8)))  float  v8f;
typedef __attribute__((ext_vector_type(4)))  unsigned int u32x4;
typedef __attribute__((ext_vector_type(4)))  int  i32x4;
typedef __attribute__((ext_vector_type(8)))  int  i32x8;

#define SHFL16(lo, hi) __builtin_shufflevector(lo, hi, 0,1,2,3,4,5,6,7,8,9,10,11,12,13,14,15)

#ifdef __has_builtin
#if __has_builtin(__builtin_amdgcn_tensor_load_to_lds) && __has_builtin(__builtin_amdgcn_s_wait_tensorcnt)
#define HAVE_TDM 1
#endif
#endif
#ifndef HAVE_TDM
#define HAVE_TDM 0
#endif

// ======================================================================
// Pass 1: MXFP4 fake quant-dequant of w[K,N] (blocks of 32 along N share a
// power-of-two scale), written TRANSPOSED as wt[N,K] bf16.
// ======================================================================
__global__ __launch_bounds__(256) void mxfp4_dq_tr_kernel(
    const __bf16* __restrict__ w, __bf16* __restrict__ wt, int K, int N)
{
    __shared__ __bf16 tile[128 * 72];

    const int t  = threadIdx.x;
    const int kt = blockIdx.x * 64;
    const int nt = blockIdx.y * 128;
    const int k  = t >> 2;
    const int nb = t & 3;

    const __bf16* src = w + (size_t)(kt + k) * N + nt + nb * 32;

    float f[32];
#pragma unroll
    for (int c = 0; c < 4; ++c) {
        v8bf v = *(const v8bf*)(src + c * 8);
#pragma unroll
        for (int j = 0; j < 8; ++j) f[c * 8 + j] = (float)v[j];
    }

    float amax = 0.0f;
#pragma unroll
    for (int j = 0; j < 32; ++j) amax = fmaxf(amax, fabsf(f[j]));

    const int eb = (int)((__float_as_uint(amax) >> 23) & 0xff) - 127;
    int ec = eb - 2;
    ec = ec < -127 ? -127 : (ec > 127 ? 127 : ec);
    const bool nz    = amax > 0.0f;
    const float scl  = nz ? ldexpf(1.0f, ec)  : 1.0f;
    const float iscl = nz ? ldexpf(1.0f, -ec) : 1.0f;

    const float mids[7] = {0.25f, 0.75f, 1.25f, 1.75f, 2.5f, 3.5f, 5.0f};

#pragma unroll
    for (int j = 0; j < 32; ++j) {
        float q = f[j] * iscl;
        float a = fminf(fabsf(q), 6.0f);
        int  cnt = 0;
        bool tie = false;
#pragma unroll
        for (int i = 0; i < 7; ++i) {
            cnt += (a > mids[i]) ? 1 : 0;
            tie  = tie || (a == mids[i]);
        }
        int idx = (tie && (cnt & 1)) ? cnt + 1 : cnt;
        float g = 0.0f;
        g = (idx >= 1) ? 0.5f : g;
        g = (idx >= 2) ? 1.0f : g;
        g = (idx >= 3) ? 1.5f : g;
        g = (idx >= 4) ? 2.0f : g;
        g = (idx >= 5) ? 3.0f : g;
        g = (idx >= 6) ? 4.0f : g;
        g = (idx >= 7) ? 6.0f : g;
        float d = copysignf(g * scl, q);
        tile[(nb * 32 + j) * 72 + k] = (__bf16)d;
    }

    __syncthreads();

#pragma unroll
    for (int i = 0; i < 4; ++i) {
        int c  = t + 256 * i;
        int n  = c >> 3;
        int kc = (c & 7) * 8;
        v8bf v = *(const v8bf*)&tile[n * 72 + kc];
        *(v8bf*)(wt + (size_t)(nt + n) * K + kt + kc) = v;
    }
}

// ======================================================================
// Pass 2: bf16 WMMA GEMM, out = x @ wt^T + bias.
// Double-buffered LDS tiles filled by the Tensor Data Mover (TDM):
// wave0 DMAs the A tile, wave1 the B tile for the NEXT k-step while all
// 8 waves run WMMAs on the current one; sync = s_wait_tensorcnt + barrier.
// TDM pad: rows of 64 elem (=32 DWORDs, pad_interval=4) padded by 4 DWORDs
// (pad_amount=3) -> LDS row stride 72 elements (144 B), same as SW layout.
// ======================================================================
#define BM   128
#define BN   128
#define BK   64
#define LDT  72
#define TILEB (128 * LDT)   // elements per tile buffer

#if HAVE_TDM
__device__ __forceinline__ void tdm_tile_load(const __bf16* gsrc, unsigned ldsByteOff,
                                              int kElems, int nRows)
{
    unsigned long long ga = (unsigned long long)(size_t)gsrc;
    u32x4 g0;
    g0[0] = 1u;                                        // count=1 (valid), no gather
    g0[1] = ldsByteOff;                                // lds_addr
    g0[2] = (unsigned)(ga & 0xffffffffu);              // global_addr[31:0]
    g0[3] = (unsigned)((ga >> 32) & 0x01ffffffu)       // global_addr[56:32]
          | (2u << 30);                                // type=2 (image)
    i32x8 g1;
    g1[0] = (1 << 16)                                  // data_size = 2 bytes
          | (1 << 20)                                  // pad_enable
          | (4 << 22)                                  // pad_interval: 32 DWORDs (128B rows)
          | (3 << 25);                                 // pad_amount: 4 DWORDs (16B)
    g1[1] = (int)(((unsigned)kElems & 0xffffu) << 16); // tensor_dim0[15:0] | atomic_addr=0
    g1[2] = (int)((((unsigned)kElems >> 16) & 0xffffu)
          | (((unsigned)nRows & 0xffffu) << 16));      // tensor_dim0 hi | tensor_dim1 lo
    g1[3] = (int)(((unsigned)nRows >> 16) & 0xffffu)
          | (BK << 16);                                // tensor_dim1 hi | tile_dim0=64
    g1[4] = 128;                                       // tile_dim1=128, tile_dim2=0
    g1[5] = kElems;                                    // tensor_dim0_stride[31:0]
    g1[6] = 0;                                         // stride hi, dim1_stride lo
    g1[7] = 0;                                         // dim1_stride hi
    const i32x4 z4 = {0, 0, 0, 0};
#if __clang_major__ >= 23
    const i32x8 z8 = {0, 0, 0, 0, 0, 0, 0, 0};
    __builtin_amdgcn_tensor_load_to_lds(g0, g1, z4, z4, z8, 0);
#else
    __builtin_amdgcn_tensor_load_to_lds(g0, g1, z4, z4, 0);
#endif
}
#endif

__global__ __launch_bounds__(256) void bf16_wmma_gemm_bias_kernel(
    const __bf16* __restrict__ x,    // [M,K]
    const __bf16* __restrict__ wt,   // [N,K]
    const __bf16* __restrict__ bias, // [N]
    __bf16* __restrict__ out,        // [M,N]
    int M, int N, int K)
{
    __shared__ __bf16 smem[4 * TILEB];  // buf0:{A,B} buf1:{A,B}

    const int t    = threadIdx.x;
    const int lane = t & 31;
    const int wid  = t >> 5;
    const int wm   = wid >> 2;
    const int wn   = wid & 3;
    const int l15  = lane & 15;
    const int lhi  = lane >> 4;

    const int m0 = blockIdx.y * BM;
    const int n0 = blockIdx.x * BN;

    float biasv[2];
#pragma unroll
    for (int ni = 0; ni < 2; ++ni)
        biasv[ni] = (float)bias[n0 + wn * 32 + ni * 16 + l15];

    v8f acc[4][2];
#pragma unroll
    for (int mi = 0; mi < 4; ++mi)
#pragma unroll
        for (int ni = 0; ni < 2; ++ni)
            acc[mi][ni] = (v8f){0.f, 0.f, 0.f, 0.f, 0.f, 0.f, 0.f, 0.f};

    // 8 WMMA tiles per wave per 32-wide k-step on the given buffers
    auto compute_tile = [&](const __bf16* aP, const __bf16* bP) {
#pragma unroll
        for (int kk = 0; kk < BK; kk += 32) {
            v16bf afrag[4], bfrag[2];
#pragma unroll
            for (int mi = 0; mi < 4; ++mi) {
                const int r = wm * 64 + mi * 16 + l15;
                const __bf16* p = &aP[r * LDT + kk + lhi * 8];
                v8bf lo = *(const v8bf*)p;
                v8bf hi = *(const v8bf*)(p + 16);
                afrag[mi] = SHFL16(lo, hi);
            }
#pragma unroll
            for (int ni = 0; ni < 2; ++ni) {
                const int cn = wn * 32 + ni * 16 + l15;
                const __bf16* p = &bP[cn * LDT + kk + lhi * 8];
                v8bf lo = *(const v8bf*)p;
                v8bf hi = *(const v8bf*)(p + 16);
                bfrag[ni] = SHFL16(lo, hi);
            }
#pragma unroll
            for (int mi = 0; mi < 4; ++mi)
#pragma unroll
                for (int ni = 0; ni < 2; ++ni)
                    acc[mi][ni] = __builtin_amdgcn_wmma_f32_16x16x32_bf16(
                        false, afrag[mi], false, bfrag[ni],
                        (short)0, acc[mi][ni], false, false);
        }
    };

#if HAVE_TDM
    // Scalar LDS base (runtime addrspacecast; no LDS-derived aggregate
    // initializers -- those end up as unsupported static initializers).
    const unsigned smemBase = (unsigned)(unsigned long long)(void*)&smem[0];

    // prologue: DMA tile 0 into buffer 0 (A at +0, B at +TILEB*2 bytes)
    if (wid == 0)      tdm_tile_load(x  + (size_t)m0 * K, smemBase, K, M);
    else if (wid == 1) tdm_tile_load(wt + (size_t)n0 * K, smemBase + 2u * TILEB, K, N);
    if (wid < 2) __builtin_amdgcn_s_wait_tensorcnt(0);
    __syncthreads();

    int cur = 0;
    for (int kt = 0; kt < K; kt += BK) {
        const int nxt = cur ^ 1;
        if (kt + BK < K) {
            const unsigned aOff = smemBase + (unsigned)nxt * (4u * TILEB); // bytes
            if (wid == 0)      tdm_tile_load(x  + (size_t)m0 * K + kt + BK, aOff, K, M);
            else if (wid == 1) tdm_tile_load(wt + (size_t)n0 * K + kt + BK, aOff + 2u * TILEB, K, N);
        }
        {
            const __bf16* aP = smem + cur * (2 * TILEB);
            compute_tile(aP, aP + TILEB);
        }
        if (wid < 2) __builtin_amdgcn_s_wait_tensorcnt(0);
        __syncthreads();
        cur = nxt;
    }
#else
    // fallback: single-buffered manual copy
    const int row = t >> 3;
    const int kc8 = (t & 7) * 8;
    const __bf16* aG = x  + (size_t)(m0 + row) * K + kc8;
    const __bf16* bG = wt + (size_t)(n0 + row) * K + kc8;
    for (int kt = 0; kt < K; kt += BK) {
#pragma unroll
        for (int i = 0; i < 4; ++i) {
            const int r = row + 32 * i;
            v8bf av = *(const v8bf*)(aG + (size_t)(32 * i) * K + kt);
            v8bf bv = *(const v8bf*)(bG + (size_t)(32 * i) * K + kt);
            *(v8bf*)&smem[r * LDT + kc8] = av;
            *(v8bf*)&smem[TILEB + r * LDT + kc8] = bv;
        }
        __syncthreads();
        compute_tile(smem, smem + TILEB);
        __syncthreads();
    }
#endif

    // epilogue: C/D layout -> lane = col (l&15), VGPR r = row (l>>4)*8 + r
#pragma unroll
    for (int mi = 0; mi < 4; ++mi) {
#pragma unroll
        for (int ni = 0; ni < 2; ++ni) {
            const int colg = n0 + wn * 32 + ni * 16 + l15;
#pragma unroll
            for (int r = 0; r < 8; ++r) {
                const int rowg = m0 + wm * 64 + mi * 16 + lhi * 8 + r;
                out[(size_t)rowg * N + colg] = (__bf16)(acc[mi][ni][r] + biasv[ni]);
            }
        }
    }
}

// ======================================================================
extern "C" void kernel_launch(void* const* d_in, const int* in_sizes, int n_in,
                              void* d_out, int out_size, void* d_ws, size_t ws_size,
                              hipStream_t stream) {
    const __bf16* x    = (const __bf16*)d_in[0];
    const __bf16* w    = (const __bf16*)d_in[1];
    const __bf16* bias = (const __bf16*)d_in[2];
    __bf16* out = (__bf16*)d_out;

    const int N = in_sizes[2];
    const int K = in_sizes[1] / N;
    const int M = in_sizes[0] / K;

    __bf16* wt = (__bf16*)d_ws;   // [N,K] dequantized+transposed (K*N*2 bytes)

    dim3 gq(K / 64, N / 128);
    mxfp4_dq_tr_kernel<<<gq, 256, 0, stream>>>(w, wt, K, N);

    dim3 gg(N / BN, M / BM);
    bf16_wmma_gemm_bias_kernel<<<gg, 256, 0, stream>>>(x, wt, bias, out, M, N, K);
}
